// CR8_reg_cond_mul_5_13975823582042
// MI455X (gfx1250) — matrix operand order repaired and checked by reading, not compiled, not run on hardware
//
#include <hip/hip_runtime.h>
#include <hip/hip_bf16.h>

// CDNA5 / gfx1250 fused kernel:
//   prep:  fold bn, convert all weights to bf16 once into d_ws
//   main:  4x (128x128) GEMM + argmax + (256x256) GEMM + gathered regression,
//          bf16 WMMA (f32 accum), weights staged global->LDS with
//          GLOBAL_LOAD_ASYNC_TO_LDS_B128 double-buffering (ASYNCcnt).

typedef __attribute__((ext_vector_type(16))) __bf16 v16bf;
typedef __attribute__((ext_vector_type(8)))  float  v8f;

#define NPIX   131072       // 16 * 8192
#define EPSV   1e-5f

// ---- workspace layout (bytes) ----
//   0      : Wcl1' bf16[16384]   (bn-folded)
//   32768  : Wreg1' bf16[16384]  (bn-folded)
//   65536  : Wcl2  bf16[16384]
//   98304  : Wcl3  bf16[16384]   (rows 0..127)
//   131072 : W2'   bf16[65536]   (256x256, row = s*32+o, col = f)
//   262144 : FB    f32[5136]:
//            [0:128) bias1  [128:256) biasR  [256:384) cl2_b  [384:512) cl3_b
//            [512:640) maskw(cl3 row128)     [640:896) b2 flat
//            [896:1024) b3  [1024:5120) w3   [5120] cl3_b[128]
#define WS_W1    0
#define WS_WR    32768
#define WS_W2C   65536
#define WS_W3C   98304
#define WS_W2P   131072
#define WS_FB    262144
#define FB_FLOATS 5136                       // padded to 16B multiple
#define FB_BYTES (FB_FLOATS * 4)

// ---- LDS layout (dynamic) ----
//   WBUF : bf16[65536] (131072 B); buf0 = first 16384, buf1 = next 16384
//   FBL  : f32[5136]   ( 20544 B)
#define SMEM_BYTES (131072 + FB_BYTES)

__device__ __forceinline__ float lrelu(float v) { return v >= 0.f ? v : 0.01f * v; }

// Low 32 bits of a flat pointer into LDS == wave-relative LDS byte offset
// (ISA 10.2, LDS aperture).
__device__ __forceinline__ unsigned lds_off(const void* p) {
  return (unsigned)(unsigned long long)p;
}

// Async DMA: 16 bytes global -> LDS per active lane. Tracked by ASYNCcnt.
__device__ __forceinline__ void async_copy_b128(unsigned lds_byte_off,
                                                const void* gsrc) {
  asm volatile("global_load_async_to_lds_b128 %0, %1, off"
               :: "v"(lds_byte_off),
                  "v"((unsigned long long)(size_t)gsrc)
               : "memory");
}

__device__ __forceinline__ void async_stage(unsigned lds_base, const unsigned char* src,
                                            int bytes, int tid) {
  for (int i = tid * 16; i < bytes; i += 256 * 16)
    async_copy_b128(lds_base + (unsigned)i, src + i);
}

__device__ __forceinline__ void wait_async0() {
#if __has_builtin(__builtin_amdgcn_s_wait_asynccnt)
  __builtin_amdgcn_s_wait_asynccnt(0);
#else
  asm volatile("s_wait_asynccnt 0" ::: "memory");
#endif
}

// A-fragment (16x32 bf16): lanes 0-15 = rows, K {kb..kb+15};
// lanes 16-31 same rows, K {kb+16..kb+31}.
__device__ __forceinline__ v16bf load_a_frag(const __bf16* W, int ldw, int tile,
                                             int kbase, int lane) {
  const int m     = lane & 15;
  const int hioff = (lane >= 16) ? 16 : 0;
  const __bf16* r = W + (size_t)(tile * 16 + m) * ldw + kbase + hioff;
  v16bf a;
#pragma unroll
  for (int i = 0; i < 8; i++) a[i] = r[i];
#pragma unroll
  for (int i = 0; i < 8; i++) a[8 + i] = r[8 + i];
  return a;
}

// B-fragments (K x 16 px) from global fp32 x (channel-major, stride 8192).
__device__ __forceinline__ void xfrags_global(const float* xb, int wpix, int lane,
                                              v16bf* bf) {
  const int col   = wpix + (lane & 15);
  const int koff0 = (lane >= 16) ? 16 : 0;
#pragma unroll
  for (int f = 0; f < 4; f++) {
    const float* p = xb + (size_t)(32 * f + koff0) * 8192 + col;
    v16bf b;
#pragma unroll
    for (int i = 0; i < 16; i++) b[i] = (__bf16)p[(size_t)i * 8192];
    bf[f] = b;
  }
}

// f32 accumulator tiles -> bf16 B-fragments (cross-half shfl_xor 16).
__device__ __forceinline__ void frags_from_acc(const v8f* acc, v16bf* bf, int lane) {
  const bool lo = lane < 16;
#pragma unroll
  for (int f = 0; f < 4; f++) {
    v16bf b;
#pragma unroll
    for (int j = 0; j < 8; j++) {
      float t0 = acc[2 * f][j];
      float t1 = acc[2 * f + 1][j];
      float p0 = __shfl_xor(t0, 16, 32);
      float p1 = __shfl_xor(t1, 16, 32);
      b[j]     = (__bf16)(lo ? t0 : p1);
      b[8 + j] = (__bf16)(lo ? p0 : t1);
    }
    bf[f] = b;
  }
}

template <int KT>
__device__ __forceinline__ void gemm8(const __bf16* W, int ldw, const v16bf* bf,
                                      v8f* acc, int lane) {
#pragma unroll
  for (int t = 0; t < 8; t++) {
    v8f c{0.f, 0.f, 0.f, 0.f, 0.f, 0.f, 0.f, 0.f};
#pragma unroll
    for (int f = 0; f < KT; f++) {
      v16bf a = load_a_frag(W, ldw, t, 32 * f, lane);
      c = __builtin_amdgcn_wmma_f32_16x16x32_bf16(false, a, false, bf[f],
                                                  (short)0, c, false, false);
    }
    acc[t] = c;
  }
}

__device__ __forceinline__ void bias_act(v8f* acc, const float* bias, int lane,
                                         bool act) {
  const int hioff = (lane >= 16) ? 8 : 0;
#pragma unroll
  for (int t = 0; t < 8; t++)
#pragma unroll
    for (int j = 0; j < 8; j++) {
      float v = acc[t][j] + bias[16 * t + j + hioff];
      acc[t][j] = act ? lrelu(v) : v;
    }
}

// ---------------------------------------------------------------------------
__global__ void __launch_bounds__(256)
CR8_prep_weights(const float* __restrict__ cl1_w, const float* __restrict__ cl1_b,
                 const float* __restrict__ g1, const float* __restrict__ bt1,
                 const float* __restrict__ m1, const float* __restrict__ v1,
                 const float* __restrict__ cl2_w, const float* __restrict__ cl2_b,
                 const float* __restrict__ cl3_w, const float* __restrict__ cl3_b,
                 const float* __restrict__ reg1_w, const float* __restrict__ reg1_b,
                 const float* __restrict__ g2, const float* __restrict__ bt2,
                 const float* __restrict__ m2, const float* __restrict__ v2,
                 const float* __restrict__ w2, const float* __restrict__ b2,
                 const float* __restrict__ w3, const float* __restrict__ b3,
                 unsigned char* __restrict__ ws) {
  __bf16* W1  = (__bf16*)(ws + WS_W1);
  __bf16* WR  = (__bf16*)(ws + WS_WR);
  __bf16* W2c = (__bf16*)(ws + WS_W2C);
  __bf16* W3c = (__bf16*)(ws + WS_W3C);
  __bf16* W2P = (__bf16*)(ws + WS_W2P);
  float*  FB  = (float*)(ws + WS_FB);

  const int tid  = blockIdx.x * 256 + threadIdx.x;
  const int nthr = gridDim.x * 256;

  for (int i = tid; i < 16384; i += nthr) {
    int o = i >> 7;
    float s1 = g1[o] * rsqrtf(v1[o] + EPSV);
    float s2 = g2[o] * rsqrtf(v2[o] + EPSV);
    W1[i]  = (__bf16)(cl1_w[i] * s1);
    WR[i]  = (__bf16)(reg1_w[i] * s2);
    W2c[i] = (__bf16)cl2_w[i];
    W3c[i] = (__bf16)cl3_w[i];
  }
  for (int i = tid; i < 65536; i += nthr) {
    int s = i >> 13, f = (i >> 5) & 255, o = i & 31;   // w2 flat: s*8192+f*32+o
    W2P[(s * 32 + o) * 256 + f] = (__bf16)w2[i];
  }
  for (int i = tid; i < FB_FLOATS; i += nthr) {
    float v = 0.f;
    if (i < 128) {
      float s = g1[i] * rsqrtf(v1[i] + EPSV);
      v = (cl1_b[i] - m1[i]) * s + bt1[i];
    } else if (i < 256) {
      int o = i - 128;
      float s = g2[o] * rsqrtf(v2[o] + EPSV);
      v = (reg1_b[o] - m2[o]) * s + bt2[o];
    } else if (i < 384)  v = cl2_b[i - 256];
    else if (i < 512)    v = cl3_b[i - 384];
    else if (i < 640)    v = cl3_w[128 * 128 + (i - 512)];
    else if (i < 896)    v = b2[i - 640];
    else if (i < 1024)   v = b3[i - 896];
    else if (i < 5120)   v = w3[i - 1024];
    else if (i == 5120)  v = cl3_b[128];
    FB[i] = v;
  }
}

// ---------------------------------------------------------------------------
__global__ void __launch_bounds__(256)
CR8_fused_kernel(const float* __restrict__ x_in,
                 const unsigned char* __restrict__ ws,
                 float* __restrict__ out) {
  extern __shared__ unsigned char smem[];
  __bf16* WBUF = (__bf16*)smem;
  __bf16* buf0 = WBUF;
  __bf16* buf1 = WBUF + 16384;
  float*  FBL  = (float*)(smem + 131072);
  const float* BIAS_A = FBL;
  const float* BIAS_R = FBL + 128;
  const float* BIAS_2 = FBL + 256;
  const float* BIAS_3 = FBL + 384;
  const float* MASKW  = FBL + 512;
  const float* B2F    = FBL + 640;
  const float* B3S    = FBL + 896;
  const float* W3S    = FBL + 1024;

  const int tid   = threadIdx.x;
  const int lane  = tid & 31;
  const int wv    = tid >> 5;
  const int P0    = blockIdx.x * 128 + wv * 16;
  const int batch = P0 >> 13;
  const int wpix  = P0 & 8191;
  const float* xb = x_in + (size_t)batch * 128 * 8192;

  const unsigned off_b0  = lds_off(buf0);
  const unsigned off_b1  = lds_off(buf1);
  const unsigned off_wb  = lds_off(WBUF);
  const unsigned off_fbl = lds_off(FBL);

  // Stage tables + phase-A weights (async DMA, no VGPR round trip).
  async_stage(off_fbl, ws + WS_FB, FB_BYTES, tid);
  async_stage(off_b0,  ws + WS_W1, 32768, tid);
  wait_async0();
  __syncthreads();

  v16bf xf[4];
  xfrags_global(xb, wpix, lane, xf);

  // -- Phase A: h = lrelu(bn(cl1(x)))  [stage reg1 weights concurrently] ----
  async_stage(off_b1, ws + WS_WR, 32768, tid);
  v8f acc[8];
  gemm8<4>(buf0, 128, xf, acc, lane);
  bias_act(acc, BIAS_A, lane, true);
  v16bf hf[4];
  frags_from_acc(acc, hf, lane);
  wait_async0();
  __syncthreads();

  // -- Phase B: r = lrelu(bn(reg1(x)))  [stage cl2 concurrently] ------------
  async_stage(off_b0, ws + WS_W2C, 32768, tid);
  gemm8<4>(buf1, 128, xf, acc, lane);
  bias_act(acc, BIAS_R, lane, true);
  v16bf rf[4];
  frags_from_acc(acc, rf, lane);
  wait_async0();
  __syncthreads();

  // -- Phase C: x2 = lrelu(cl2(h))  [stage cl3 concurrently] ----------------
  async_stage(off_b1, ws + WS_W3C, 32768, tid);
  gemm8<4>(buf0, 128, hf, acc, lane);
  bias_act(acc, BIAS_2, lane, true);
  v16bf x2f[4];
  frags_from_acc(acc, x2f, lane);
  wait_async0();
  __syncthreads();

  // -- Phase D: logits = cl3(x2), rows 0..127 -------------------------------
  gemm8<4>(buf1, 128, x2f, acc, lane);
  bias_act(acc, BIAS_3, lane, false);

  // argmax over 128 classes per pixel (first-max tie break).
  const int hioff = (lane >= 16) ? 8 : 0;
  float bv = -3.0e38f;
  int   bi = 0;
#pragma unroll
  for (int t = 0; t < 8; t++)
#pragma unroll
    for (int j = 0; j < 8; j++) {
      float v = acc[t][j];
      int ch  = 16 * t + j + hioff;
      if (v > bv || (v == bv && ch < bi)) { bv = v; bi = ch; }
    }
  {
    float pv = __shfl_xor(bv, 16, 32);
    int   pi = __shfl_xor(bi, 16, 32);
    if (pv > bv || (pv == bv && pi < bi)) { bv = pv; bi = pi; }
  }
  const int ind = bi;
  const int sup = ind >> 4;   // CLASS_FACTOR = 16

  // mask = lrelu(cl3 row 128 of x2) via VALU dot on x2 fragments.
  float mpart = 0.f;
  const int koff0 = (lane >= 16) ? 16 : 0;
#pragma unroll
  for (int f = 0; f < 4; f++)
#pragma unroll
    for (int i = 0; i < 16; i++)
      mpart += (float)x2f[f][i] * MASKW[32 * f + koff0 + i];
  mpart += __shfl_xor(mpart, 16, 32);
  const float maskv = lrelu(mpart + FBL[5120]);

  // -- Phase E: y_all = W2'(256x256) * cat([r;h]) ---------------------------
  __syncthreads();                               // all waves done with buf0/buf1
  async_stage(off_wb, ws + WS_W2P, 131072, tid);
  wait_async0();
  __syncthreads();

  v16bf cf[8];
#pragma unroll
  for (int f = 0; f < 4; f++) { cf[f] = rf[f]; cf[4 + f] = hf[f]; }
#pragma unroll
  for (int t = 0; t < 8; t++) {
    v8f c{0.f, 0.f, 0.f, 0.f, 0.f, 0.f, 0.f, 0.f};
#pragma unroll
    for (int f = 0; f < 8; f++) {
      v16bf a = load_a_frag(WBUF, 256, t, 32 * f, lane);
      c = __builtin_amdgcn_wmma_f32_16x16x32_bf16(false, a, false, cf[f],
                                                  (short)0, c, false, false);
    }
    acc[t] = c;
  }

  // Gather selected super block, lrelu, dot with w3[ind], + b3[ind].
  float reg = 0.f;
#pragma unroll
  for (int t = 0; t < 8; t++) {
    if ((t >> 1) == sup) {
#pragma unroll
      for (int j = 0; j < 8; j++) {
        int r   = 16 * t + j + hioff;             // row = sup*32 + o
        float y = lrelu(acc[t][j] + B2F[r]);
        reg += y * W3S[ind * 32 + (r & 31)];
      }
    }
  }
  reg += __shfl_xor(reg, 16, 32);
  reg += B3S[ind];

  if (lane < 16) {
    int p = P0 + lane;
    out[p]        = ((float)ind + reg) * (1.0f / 128.0f);
    out[NPIX + p] = maskv;
  }
}

// ---------------------------------------------------------------------------
extern "C" void kernel_launch(void* const* d_in, const int* in_sizes, int n_in,
                              void* d_out, int out_size, void* d_ws, size_t ws_size,
                              hipStream_t stream) {
  (void)in_sizes; (void)n_in; (void)ws_size; (void)out_size;

  const float* x_in   = (const float*)d_in[0];
  const float* cl1_w  = (const float*)d_in[1];
  const float* cl1_b  = (const float*)d_in[2];
  const float* g1     = (const float*)d_in[3];
  const float* bt1    = (const float*)d_in[4];
  const float* m1     = (const float*)d_in[5];
  const float* v1     = (const float*)d_in[6];
  const float* cl2_w  = (const float*)d_in[7];
  const float* cl2_b  = (const float*)d_in[8];
  const float* cl3_w  = (const float*)d_in[9];
  const float* cl3_b  = (const float*)d_in[10];
  const float* reg1_w = (const float*)d_in[11];
  const float* reg1_b = (const float*)d_in[12];
  const float* g2     = (const float*)d_in[13];
  const float* bt2    = (const float*)d_in[14];
  const float* m2     = (const float*)d_in[15];
  const float* v2     = (const float*)d_in[16];
  const float* w2     = (const float*)d_in[17];
  const float* b2     = (const float*)d_in[18];
  const float* w3     = (const float*)d_in[19];
  const float* b3     = (const float*)d_in[20];
  float* out = (float*)d_out;
  unsigned char* ws = (unsigned char*)d_ws;

  CR8_prep_weights<<<dim3(64), dim3(256), 0, stream>>>(
      cl1_w, cl1_b, g1, bt1, m1, v1, cl2_w, cl2_b, cl3_w, cl3_b,
      reg1_w, reg1_b, g2, bt2, m2, v2, w2, b2, w3, b3, ws);

  (void)hipFuncSetAttribute((const void*)CR8_fused_kernel,
                            hipFuncAttributeMaxDynamicSharedMemorySize,
                            SMEM_BYTES);

  CR8_fused_kernel<<<dim3(NPIX / 128), dim3(256), SMEM_BYTES, stream>>>(
      x_in, ws, out);
}
